// QKVStableMaskedAttention_67353677136135
// MI455X (gfx1250) — matrix operand order repaired and checked by reading, not compiled, not run on hardware
//
#include <hip/hip_runtime.h>
#include <math.h>

// ---------------------------------------------------------------------------
// Fused masked multi-head attention for MI455X (gfx1250, wave32, WMMA).
//   q,k,v : (2, 1024, 2048) fp32  -> 32 heads x (ch=64, L=2048)
//   mask  : (2, 2048, 2048) bool
//   out   : a (2,1024,2048) fp32  ++  w (32,2048,2048) fp32   [tuple, concat]
// Memory-bound on the 512MB w store (~26us @ 23.3TB/s). Compute via
// v_wmma_f32_16x16x32_bf16 (fp32 accumulate). Two-phase softmax (branchless
// online stats pass + emit pass). Masked scores use a -1e30 sentinel so the
// whole inner loop is exec-mask-free (no cmpx/branch/v_nop storms).
// ---------------------------------------------------------------------------

#define L       2048
#define CH      64
#define TQ      128
#define TK      128
#define WAVES   8
#define NEG_BIG (-1.0e30f)

typedef __attribute__((ext_vector_type(16))) __bf16 v16bf;
typedef __attribute__((ext_vector_type(8)))  float  v8f;

__device__ __forceinline__ unsigned short f2bf(float f) {
  unsigned int u = __float_as_uint(f);
  unsigned int r = (u + 0x7FFFu + ((u >> 16) & 1u)) >> 16;   // round-nearest-even
  return (unsigned short)r;
}

// 16 contiguous bf16 (B-matrix operand: lane's K-run for its column)
__device__ __forceinline__ v16bf ld16(const unsigned short* p) {
  v16bf r;
  uint4* d = reinterpret_cast<uint4*>(&r);
  d[0] = *reinterpret_cast<const uint4*>(p);
  d[1] = *reinterpret_cast<const uint4*>(p + 8);
  return r;
}

// A-matrix operand (contiguous K order, same as ld16 B): lanes 0-15 take K {0..15},
// lanes 16-31 take K {16..31} of this lane's row.
__device__ __forceinline__ v16bf ldA(const unsigned short* row, int cb, int hi) {
  v16bf r;
  uint4* d = reinterpret_cast<uint4*>(&r);
  d[0] = *reinterpret_cast<const uint4*>(row + cb + hi * 16);
  d[1] = *reinterpret_cast<const uint4*>(row + cb + hi * 16 + 8);
  return r;
}

__global__ __launch_bounds__(256, 1)
void attn_fused_kernel(const float* __restrict__ q, const float* __restrict__ k,
                       const float* __restrict__ v, const unsigned char* __restrict__ mask,
                       float* __restrict__ out_a, float* __restrict__ out_w) {
  // LDS staging (63KB total). Row strides padded & kept 16B-aligned.
  __shared__ __align__(16) unsigned short Qs[TQ][72];        // [q][c]  (transposed)
  __shared__ __align__(16) unsigned short Ks[TK][72];        // [k][c]  (transposed)
  __shared__ __align__(16) unsigned short Vs[CH][136];       // [c][k]  (natural)
  __shared__ __align__(16) unsigned short Ps[WAVES][16][40]; // per-wave P scratch

  const int tid  = threadIdx.x;
  const int lane = tid & 31;
  const int wid  = tid >> 5;
  const int hi   = lane >> 4;      // 0: lanes 0-15, 1: lanes 16-31
  const int ln   = lane & 15;

  const int n  = blockIdx.x >> 4;          // head-batch index, 0..31
  const int q0 = (blockIdx.x & 15) * TQ;   // q tile origin
  const int b  = n >> 4;                   // batch
  const int hh = n & 15;                   // head within batch
  const int qrow = wid * 16;               // this wave's local q rows

  const size_t headoff = (size_t)n * CH * L;
  const float* qg = q + headoff;
  const float* kg = k + headoff;
  const float* vg = v + headoff;
  const unsigned char* mg = mask + (size_t)b * L * L;

  // ---- stage Q tile once, pre-scaled by 1/sqrt(64)=0.125 (exact pow2):
  //      Qs[qi][c] = bf16(0.125 * q[n][c][q0+qi])   (coalesced global reads)
  for (int idx = tid; idx < CH * TQ; idx += 256) {
    int c = idx >> 7, qi = idx & 127;
    Qs[qi][c] = f2bf(0.125f * qg[(size_t)c * L + q0 + qi]);
  }
  __syncthreads();

  // Q A-operands are invariant for the whole kernel -> hoist into registers
  // (compiler can't LICM LDS loads across barriers).
  const unsigned short* qrow_p = &Qs[qrow + ln][0];
  const v16bf QA0 = ldA(qrow_p, 0,  hi);
  const v16bf QA1 = ldA(qrow_p, 32, hi);

  // =========================================================================
  // Phase 1: branchless per-row online softmax stats (max m, sum s).
  // =========================================================================
  float rm[8], rs[8];
#pragma unroll
  for (int j = 0; j < 8; ++j) { rm[j] = NEG_BIG; rs[j] = 0.0f; }

  for (int kc = 0; kc < L; kc += TK) {
    __syncthreads();
    for (int idx = tid; idx < CH * TK; idx += 256) {
      int c = idx >> 7, ki = idx & 127;
      Ks[ki][c] = f2bf(kg[(size_t)c * L + kc + ki]);
    }
    __syncthreads();

#pragma unroll
    for (int kb = 0; kb < TK / 16; ++kb) {
      v8f s = {};
      {
        v16bf B0 = ld16(&Ks[kb * 16 + ln][0  + hi * 16]);
        v16bf B1 = ld16(&Ks[kb * 16 + ln][32 + hi * 16]);
        s = __builtin_amdgcn_wmma_f32_16x16x32_bf16(false, QA0, false, B0,
                                                    (short)0, s, false, false);
        s = __builtin_amdgcn_wmma_f32_16x16x32_bf16(false, QA1, false, B1,
                                                    (short)0, s, false, false);
      }
      const int kcol = kc + kb * 16 + ln;
      // batch the 8 mask byte loads (keeps them all in flight, single wait)
      unsigned char mb[8];
#pragma unroll
      for (int j = 0; j < 8; ++j) {
        int qa = q0 + qrow + hi * 8 + j;
        mb[j] = mg[(size_t)qa * L + kcol];
      }
#pragma unroll
      for (int j = 0; j < 8; ++j) {
        float x  = mb[j] ? s[j] : NEG_BIG;     // scale already folded into Q
        float mn = fmaxf(rm[j], x);
        rs[j] = rs[j] * __expf(rm[j] - mn) + __expf(x - mn);
        rm[j] = mn;
      }
    }
  }

  // branchless cross-lane combine over the 16 lanes that share each row
  float rinv[8];
#pragma unroll
  for (int j = 0; j < 8; ++j) {
    float m = rm[j], ss = rs[j];
#pragma unroll
    for (int off = 1; off < 16; off <<= 1) {
      float m2 = __shfl_xor(m, off, 32);
      float s2 = __shfl_xor(ss, off, 32);
      float mm = fmaxf(m, m2);
      ss = ss * __expf(m - mm) + s2 * __expf(m2 - mm);
      m = mm;
    }
    rm[j]   = m;
    rinv[j] = 1.0f / ss;
  }

  // =========================================================================
  // Phase 2: recompute scores, emit w = exp(x-m)/s, accumulate a = P * V^T.
  // =========================================================================
  v8f acc[4];
#pragma unroll
  for (int ct = 0; ct < 4; ++ct) acc[ct] = {};

  for (int kc = 0; kc < L; kc += TK) {
    __syncthreads();
    for (int idx = tid; idx < CH * TK; idx += 256) {
      int c = idx >> 7, ki = idx & 127;
      Ks[ki][c] = f2bf(kg[(size_t)c * L + kc + ki]);
      Vs[c][ki] = f2bf(vg[(size_t)c * L + kc + ki]);
    }
    __syncthreads();

#pragma unroll
    for (int kb2 = 0; kb2 < TK / 32; ++kb2) {
#pragma unroll
      for (int h = 0; h < 2; ++h) {             // two 16-col score subtiles
        const int kb = kb2 * 2 + h;
        v8f s = {};
        {
          v16bf B0 = ld16(&Ks[kb * 16 + ln][0  + hi * 16]);
          v16bf B1 = ld16(&Ks[kb * 16 + ln][32 + hi * 16]);
          s = __builtin_amdgcn_wmma_f32_16x16x32_bf16(false, QA0, false, B0,
                                                      (short)0, s, false, false);
          s = __builtin_amdgcn_wmma_f32_16x16x32_bf16(false, QA1, false, B1,
                                                      (short)0, s, false, false);
        }
        const int kcol = kc + kb * 16 + ln;
        unsigned char mb[8];
#pragma unroll
        for (int j = 0; j < 8; ++j) {
          int qa = q0 + qrow + hi * 8 + j;
          mb[j] = mg[(size_t)qa * L + kcol];
        }
#pragma unroll
        for (int j = 0; j < 8; ++j) {
          int qa  = q0 + qrow + hi * 8 + j;
          float x = mb[j] ? s[j] : NEG_BIG;
          float p = __expf(x - rm[j]) * rinv[j];
          out_w[(size_t)n * L * L + (size_t)qa * L + kcol] = p;  // 512MB stream
          Ps[wid][hi * 8 + j][h * 16 + ln] = f2bf(p);
        }
      }
      // GEMM2: acc(16q x 16c) += P(16q x 32k) * V^T(32k x 16c)
      const unsigned short* prow = &Ps[wid][ln][0];
      v16bf A = ldA(prow, 0, hi);              // same-wave LDS, in-order via dscnt
#pragma unroll
      for (int ct = 0; ct < 4; ++ct) {
        v16bf B = ld16(&Vs[ct * 16 + ln][kb2 * 32 + hi * 16]);
        acc[ct] = __builtin_amdgcn_wmma_f32_16x16x32_bf16(false, A, false, B,
                                                          (short)0, acc[ct], false, false);
      }
    }
  }

  // ---- store a: out_a[(b*1024 + hh*64 + c)*L + q]
#pragma unroll
  for (int ct = 0; ct < 4; ++ct) {
    const int c = ct * 16 + ln;
    const size_t rowbase = ((size_t)b * 1024 + (size_t)hh * 64 + c) * L;
#pragma unroll
    for (int j = 0; j < 8; ++j) {
      int qa = q0 + qrow + hi * 8 + j;
      out_a[rowbase + qa] = acc[ct][j];
    }
  }
}

extern "C" void kernel_launch(void* const* d_in, const int* in_sizes, int n_in,
                              void* d_out, int out_size, void* d_ws, size_t ws_size,
                              hipStream_t stream) {
  (void)in_sizes; (void)n_in; (void)out_size; (void)d_ws; (void)ws_size;
  const float* q = (const float*)d_in[0];
  const float* k = (const float*)d_in[1];
  const float* v = (const float*)d_in[2];
  const unsigned char* mask = (const unsigned char*)d_in[3];
  float* out_a = (float*)d_out;
  float* out_w = out_a + (size_t)2 * 1024 * 2048;   // tuple: a then w

  // 32 head-batches x 16 q-tiles of 128 rows; 256 threads = 8 wave32s.
  attn_fused_kernel<<<dim3(512), dim3(256), 0, stream>>>(q, k, v, mask, out_a, out_w);
}
